// CoAttnGateGCN_91199335563573
// MI455X (gfx1250) — matrix operand order repaired and checked by reading, not compiled, not run on hardware
//
#include <hip/hip_runtime.h>
#include <hip/hip_bf16.h>

// ---------------------------------------------------------------------------
// CDNA5 (gfx1250) implementation of CoAttnGateGCN forward.
// GEMM blocks use v_wmma_f32_16x16x32_bf16 (wave32, 16x16x32 tiles).
// ---------------------------------------------------------------------------

typedef __attribute__((ext_vector_type(16))) __bf16 v16bf;
typedef __attribute__((ext_vector_type(8)))  float  v8f;

#define NN 50000
#define EE 800000
#define GG 500
#define DD 256
#define HH 128
#define LL 2

__device__ __forceinline__ float sigmoidf_(float v) { return 1.f / (1.f + __expf(-v)); }

__device__ __forceinline__ v8f wmma_bf16(v16bf a, v16bf b, v8f c) {
  return __builtin_amdgcn_wmma_f32_16x16x32_bf16(false, a, false, b, (short)0, c, false, false);
}

// A fragment (16x32 bf16), ISA 7.12.2 layout:
// lane l: m = l&15, half = l>>4; vec elem i -> k = ((i>>3)<<4) + half*8 + (i&7)
__device__ __forceinline__ v16bf load_a_frag(const __bf16* tile, int ldk, int kbase) {
  const int lane = threadIdx.x & 31;
  const int m = lane & 15, half = lane >> 4;
  const __bf16* p = tile + m * ldk + kbase + half * 8;
  v16bf a;
#pragma unroll
  for (int i = 0; i < 8; ++i) a[i] = p[i];
#pragma unroll
  for (int i = 0; i < 8; ++i) a[8 + i] = p[16 + i];
  return a;
}

// B fragment (32x16 bf16) from transposed weights Wt[n][k]:
// lane l: n = l&15, vec elem i -> k = kbase + ((i>>3)<<4) + 8*(l>>4) + (i&7)
__device__ __forceinline__ v16bf load_b_frag(const __bf16* Wt, int K, int ncol, int kbase) {
  const int lane = threadIdx.x & 31;
  const int n = lane & 15, half = lane >> 4;
  const __bf16* p = Wt + (size_t)(ncol + n) * K + kbase + half * 8;
  v16bf b;
#pragma unroll
  for (int i = 0; i < 8; ++i) b[i] = p[i];
#pragma unroll
  for (int i = 0; i < 8; ++i) b[8 + i] = p[16 + i];
  return b;
}

// --------------------------- weight prep -----------------------------------
// src: [K][Nc] fp32 row-major  ->  dst: [Nc][K] bf16
__global__ void k_t2bf(const float* __restrict__ s, __bf16* __restrict__ d, int K, int Nc) {
  int i = blockIdx.x * 256 + threadIdx.x;
  if (i >= K * Nc) return;
  int k = i / Nc, n = i - k * Nc;
  d[(size_t)n * K + k] = (__bf16)s[i];
}

// --------------------------- modality projection ---------------------------
// h = select(mask, relu(x@We+be), relu(x@Wf+bf));  16 rows/block, 8 waves.
__global__ void k_proj(const float* __restrict__ x, const int* __restrict__ em,
                       const __bf16* __restrict__ WeT, const __bf16* __restrict__ WfT,
                       const float* __restrict__ be, const float* __restrict__ bf,
                       float* __restrict__ h) {
  __shared__ __bf16 xt[16 * 264];
  const int tid = threadIdx.x;
  const int row0 = blockIdx.x * 16;
  __builtin_prefetch(WeT, 0, 3);
  __builtin_prefetch(WfT, 0, 3);
  for (int i = tid; i < 16 * 256; i += 256) {
    int r = i >> 8, c = i & 255;
    xt[r * 264 + c] = (__bf16)x[(size_t)(row0 + r) * DD + c];
  }
  __syncthreads();
  const int wave = tid >> 5, lane = tid & 31;
  const int n = lane & 15, half = lane >> 4;
  const int ncol = wave * 16;
  v8f accE = {}; v8f accF = {};
#pragma unroll
  for (int ks = 0; ks < 8; ++ks) {
    const int kb = ks * 32;
    v16bf a = load_a_frag(xt, 264, kb);
    accE = wmma_bf16(a, load_b_frag(WeT, DD, ncol, kb), accE);
    accF = wmma_bf16(a, load_b_frag(WfT, DD, ncol, kb), accF);
  }
  const int col = ncol + n;
  const float bE = be[col], bF = bf[col];
#pragma unroll
  for (int r = 0; r < 8; ++r) {
    const int row = row0 + r + half * 8;
    const float v = em[row] ? (accE[r] + bE) : (accF[r] + bF);
    h[(size_t)row * HH + col] = fmaxf(v, 0.f);
  }
}

// --------------------------- GCN weight transform --------------------------
__global__ void k_xform(const float* __restrict__ h,
                        const __bf16* __restrict__ WiT, const __bf16* __restrict__ WcT,
                        float* __restrict__ xi, float* __restrict__ xc) {
  __shared__ __bf16 ht[16 * 136];
  const int tid = threadIdx.x;
  const int row0 = blockIdx.x * 16;
  __builtin_prefetch(WiT, 0, 3);
  __builtin_prefetch(WcT, 0, 3);
  for (int i = tid; i < 16 * 128; i += 256) {
    int r = i >> 7, c = i & 127;
    ht[r * 136 + c] = (__bf16)h[(size_t)(row0 + r) * HH + c];
  }
  __syncthreads();
  const int wave = tid >> 5, lane = tid & 31;
  const int n = lane & 15, half = lane >> 4;
  const int ncol = wave * 16;
  v8f ai = {}; v8f ac = {};
#pragma unroll
  for (int ks = 0; ks < 4; ++ks) {
    const int kb = ks * 32;
    v16bf a = load_a_frag(ht, 136, kb);
    ai = wmma_bf16(a, load_b_frag(WiT, HH, ncol, kb), ai);
    ac = wmma_bf16(a, load_b_frag(WcT, HH, ncol, kb), ac);
  }
  const int col = ncol + n;
#pragma unroll
  for (int r = 0; r < 8; ++r) {
    const int row = row0 + r + half * 8;
    xi[(size_t)row * HH + col] = ai[r];
    xc[(size_t)row * HH + col] = ac[r];
  }
}

// --------------------------- edge MLP (WMMA) --------------------------------
// 4 waves/block, 16 edges/wave. eh = relu(|hs-hd|@W1[0:128] + w*W1[128] + b1)
// g = sigmoid(eh@W2 + b2); ew_hat = w*clip(g,0.2,1.2) split intra/cross.
__global__ void k_edge(const int* __restrict__ src, const int* __restrict__ dst,
                       const float* __restrict__ ew, const int* __restrict__ em,
                       const float* __restrict__ h,
                       const __bf16* __restrict__ W1T, const float* __restrict__ W1last,
                       const float* __restrict__ b1, const float* __restrict__ W2,
                       const float* __restrict__ b2,
                       float* __restrict__ ew_i, float* __restrict__ ew_c) {
  __shared__ __bf16 dt[4][16 * 136];
  const int tid = threadIdx.x, wave = tid >> 5, lane = tid & 31;
  const int ebase = (blockIdx.x * 4 + wave) * 16;
  __bf16* tile = dt[wave];
  __builtin_prefetch(W1T, 0, 3);
  // stage |h_src - h_dst| as bf16 (4 feats/lane via float4)
  for (int e = 0; e < 16; ++e) {
    const int s = src[ebase + e], d = dst[ebase + e];
    const float4 hs = *(const float4*)(h + (size_t)s * HH + lane * 4);
    const float4 hd = *(const float4*)(h + (size_t)d * HH + lane * 4);
    const int c = lane * 4;
    tile[e * 136 + c + 0] = (__bf16)fabsf(hs.x - hd.x);
    tile[e * 136 + c + 1] = (__bf16)fabsf(hs.y - hd.y);
    tile[e * 136 + c + 2] = (__bf16)fabsf(hs.z - hd.z);
    tile[e * 136 + c + 3] = (__bf16)fabsf(hs.w - hd.w);
  }
  __syncthreads();
  const int n = lane & 15, half = lane >> 4;
  v8f acc0 = {}, acc1 = {}, acc2 = {}, acc3 = {};
#pragma unroll
  for (int ks = 0; ks < 4; ++ks) {
    const int kb = ks * 32;
    v16bf a = load_a_frag(tile, 136, kb);
    acc0 = wmma_bf16(a, load_b_frag(W1T, HH, 0,  kb), acc0);
    acc1 = wmma_bf16(a, load_b_frag(W1T, HH, 16, kb), acc1);
    acc2 = wmma_bf16(a, load_b_frag(W1T, HH, 32, kb), acc2);
    acc3 = wmma_bf16(a, load_b_frag(W1T, HH, 48, kb), acc3);
  }
  const float w1l0 = W1last[n],       w1l1 = W1last[16 + n];
  const float w1l2 = W1last[32 + n],  w1l3 = W1last[48 + n];
  const float b10 = b1[n],      b11 = b1[16 + n];
  const float b12 = b1[32 + n], b13 = b1[48 + n];
  const float w20 = W2[n],      w21 = W2[16 + n];
  const float w22 = W2[32 + n], w23 = W2[48 + n];
  const float bb2 = b2[0];
#pragma unroll
  for (int r = 0; r < 8; ++r) {
    const int e = ebase + r + half * 8;
    const float w = ew[e];
    float p = fmaxf(acc0[r] + w * w1l0 + b10, 0.f) * w20
            + fmaxf(acc1[r] + w * w1l1 + b11, 0.f) * w21
            + fmaxf(acc2[r] + w * w1l2 + b12, 0.f) * w22
            + fmaxf(acc3[r] + w * w1l3 + b13, 0.f) * w23;
    p += __shfl_xor(p, 8, 16);
    p += __shfl_xor(p, 4, 16);
    p += __shfl_xor(p, 2, 16);
    p += __shfl_xor(p, 1, 16);
    if (n == r) {  // one writer lane per row (r < 8 always)
      const float g = sigmoidf_(p + bb2);
      const float wh = w * fminf(fmaxf(g, 0.2f), 1.2f);
      const int s = src[e], d = dst[e];
      const float intra = ((em[s] != 0) == (em[d] != 0)) ? 1.f : 0.f;
      ew_i[e] = wh * intra;
      ew_c[e] = wh * (1.f - intra);
    }
  }
}

// --------------------------- degree / norm ---------------------------------
__global__ void k_fill1(float* __restrict__ p, int n) {
  int i = blockIdx.x * 256 + threadIdx.x;
  if (i < n) p[i] = 1.f;
}
__global__ void k_deg(const int* __restrict__ dst, const float* __restrict__ ei,
                      const float* __restrict__ ec, float* __restrict__ di,
                      float* __restrict__ dc) {
  int e = blockIdx.x * 256 + threadIdx.x;
  int d = dst[e];
  atomicAdd(&di[d], ei[e]);
  atomicAdd(&dc[d], ec[e]);
}
__global__ void k_rsqrt(float* __restrict__ p, int n) {
  int i = blockIdx.x * 256 + threadIdx.x;
  if (i < n) p[i] = rsqrtf(p[i]);
}

// --------------------------- edge scatter ----------------------------------
__global__ void k_scatter(const int* __restrict__ src, const int* __restrict__ dst,
                          const float* __restrict__ ei, const float* __restrict__ ec,
                          const float* __restrict__ di, const float* __restrict__ dc,
                          const float* __restrict__ xi, const float* __restrict__ xc,
                          float* __restrict__ ai, float* __restrict__ ac) {
  const int gid = blockIdx.x * 256 + threadIdx.x;
  const int e = gid >> 5, f = (gid & 31) * 4;
  const int s = src[e], d = dst[e];
  const float ni = di[s] * ei[e] * di[d];
  if (ni != 0.f) {
    const float4 v = *(const float4*)(xi + (size_t)s * HH + f);
    float* o = ai + (size_t)d * HH + f;
    atomicAdd(o + 0, v.x * ni); atomicAdd(o + 1, v.y * ni);
    atomicAdd(o + 2, v.z * ni); atomicAdd(o + 3, v.w * ni);
  }
  const float nc = dc[s] * ec[e] * dc[d];
  if (nc != 0.f) {
    const float4 v = *(const float4*)(xc + (size_t)s * HH + f);
    float* o = ac + (size_t)d * HH + f;
    atomicAdd(o + 0, v.x * nc); atomicAdd(o + 1, v.y * nc);
    atomicAdd(o + 2, v.z * nc); atomicAdd(o + 3, v.w * nc);
  }
}

// --------------------------- combine + BN stats ----------------------------
__global__ void k_combine(const float* __restrict__ xi, const float* __restrict__ xc,
                          const float* __restrict__ ai, const float* __restrict__ ac,
                          const float* __restrict__ di, const float* __restrict__ dc,
                          const float* __restrict__ bi, const float* __restrict__ bc,
                          const float* __restrict__ rel_gate, int l, int n,
                          float* __restrict__ h, float* __restrict__ bn) {
  const int col = threadIdx.x;
  const int r0 = blockIdx.x * 64;
  const float g = sigmoidf_(rel_gate[l]);
  const float biv = bi[col], bcv = bc[col];
  float s = 0.f, sq = 0.f;
  const int rend = (r0 + 64 < n) ? r0 + 64 : n;
  for (int r = r0; r < rend; ++r) {
    const size_t idx = (size_t)r * HH + col;
    const float si = di[r], sc = dc[r];
    const float v = ai[idx] + xi[idx] * si * si + biv;
    const float u = ac[idx] + xc[idx] * sc * sc + bcv;
    const float hp = v + g * u;
    h[idx] = hp;
    s += hp; sq += hp * hp;
  }
  atomicAdd(&bn[col], s);
  atomicAdd(&bn[HH + col], sq);
}
__global__ void k_bnfinal(float* __restrict__ bn, const float* __restrict__ gamma,
                          const float* __restrict__ beta, int n) {
  const int c = threadIdx.x;
  const float mu = bn[c] / n;
  const float var = bn[HH + c] / n - mu * mu;
  const float rstd = rsqrtf(var + 1e-5f);
  const float ga = gamma[c];
  bn[c] = ga * rstd;
  bn[HH + c] = beta[c] - ga * rstd * mu;
}
__global__ void k_bnapply(float* __restrict__ h, const float* __restrict__ bn) {
  const int i = blockIdx.x * 256 + threadIdx.x;
  const int c = i & 127;
  h[i] = fmaxf(bn[c] * h[i] + bn[HH + c], 0.f);
}

// --------------------------- pooling ---------------------------------------
__global__ void k_pool(const float* __restrict__ h, const int* __restrict__ batch,
                       const int* __restrict__ em, int n,
                       float* __restrict__ sg, float* __restrict__ se,
                       float* __restrict__ sf, float* __restrict__ cnt) {
  const int col = threadIdx.x;
  const int r0 = blockIdx.x * 64;
  const int rend = (r0 + 64 < n) ? r0 + 64 : n;
  for (int r = r0; r < rend; ++r) {
    const int g = batch[r];
    const int m = em[r];
    const float v = h[(size_t)r * HH + col];
    atomicAdd(&sg[(size_t)g * HH + col], v);
    if (m) atomicAdd(&se[(size_t)g * HH + col], v);
    else   atomicAdd(&sf[(size_t)g * HH + col], v);
    if (col == 0) {
      atomicAdd(&cnt[g], 1.f);
      atomicAdd(&cnt[GG + g], m ? 1.f : 0.f);
      atomicAdd(&cnt[2 * GG + g], m ? 0.f : 1.f);
    }
  }
}
__global__ void k_pool_div(float* __restrict__ sg, float* __restrict__ se,
                           float* __restrict__ sf, const float* __restrict__ cnt) {
  const int g = blockIdx.x, t = threadIdx.x;
  sg[(size_t)g * HH + t] /= fmaxf(cnt[g], 1.f);
  se[(size_t)g * HH + t] /= fmaxf(cnt[GG + g], 1.f);
  sf[(size_t)g * HH + t] /= fmaxf(cnt[2 * GG + g], 1.f);
}

// --------------------------- per-graph head --------------------------------
__global__ void k_head(const float* __restrict__ he_, const float* __restrict__ hf_,
                       const float* __restrict__ hg_,
                       const float* __restrict__ coW1, const float* __restrict__ cob1,
                       const float* __restrict__ coW2, const float* __restrict__ cob2,
                       const float* __restrict__ gW1, const float* __restrict__ gb1,
                       const float* __restrict__ gW2, const float* __restrict__ gb2,
                       const float* __restrict__ cW1, const float* __restrict__ cb1,
                       const float* __restrict__ cW2, const float* __restrict__ cb2,
                       const float* __restrict__ peW1, const float* __restrict__ peb1,
                       const float* __restrict__ peW2, const float* __restrict__ peb2,
                       const float* __restrict__ pfW1, const float* __restrict__ pfb1,
                       const float* __restrict__ pfW2, const float* __restrict__ pfb2,
                       float* __restrict__ out_logits, float* __restrict__ out_ze,
                       float* __restrict__ out_zf) {
  __shared__ float she[HH], shf[HH], shg[HH], s1[HH], s2[HH], red[HH];
  __shared__ float sA[3];
  const int t = threadIdx.x;
  const int g = blockIdx.x;
  she[t] = he_[(size_t)g * HH + t];
  shf[t] = hf_[(size_t)g * HH + t];
  shg[t] = hg_[(size_t)g * HH + t];
  __syncthreads();
  // co-MLP hidden
  float acc = cob1[t];
  for (int k = 0; k < HH; ++k) {
    acc += she[k] * coW1[(size_t)k * HH + t];
    acc += shf[k] * coW1[(size_t)(HH + k) * HH + t];
    acc += she[k] * shf[k] * coW1[(size_t)(2 * HH + k) * HH + t];
  }
  s1[t] = fmaxf(acc, 0.f);
  __syncthreads();
  if (t < 2) {
    float a = cob2[t];
    for (int k = 0; k < HH; ++k) a += s1[k] * coW2[k * 2 + t];
    sA[t] = sigmoidf_(a);
  }
  __syncthreads();
  const float asum = sA[0] + sA[1] + 1e-6f;
  const float a0 = sA[0] / asum, a1 = sA[1] / asum;
  const float hcr = a0 * she[t] + a1 * shf[t];
  __syncthreads();
  s1[t] = hcr;                           // s1 := h_cr
  __syncthreads();
  // gate MLP
  acc = gb1[t];
  for (int k = 0; k < HH; ++k) {
    acc += s1[k] * gW1[(size_t)k * HH + t];
    acc += shg[k] * gW1[(size_t)(HH + k) * HH + t];
  }
  s2[t] = fmaxf(acc, 0.f);
  __syncthreads();
  if (t == 0) {
    float a = gb2[0];
    for (int k = 0; k < HH; ++k) a += s2[k] * gW2[k];
    sA[2] = sigmoidf_(a);
  }
  __syncthreads();
  const float gte = sA[2];
  const float hfin = gte * s1[t] + (1.f - gte) * shg[t];
  __syncthreads();
  s2[t] = hfin;                          // s2 := h_final
  __syncthreads();
  // classifier
  if (t < 64) {
    float a = cb1[t];
    for (int k = 0; k < HH; ++k) a += s2[k] * cW1[k * 64 + t];
    red[t] = fmaxf(a, 0.f);
  }
  __syncthreads();
  if (t < 2) {
    float a = cb2[t];
    for (int k = 0; k < 64; ++k) a += red[k] * cW2[k * 2 + t];
    out_logits[g * 2 + t] = a;
  }
  // projection head (eeg)
  __syncthreads();
  acc = peb1[t];
  for (int k = 0; k < HH; ++k) acc += she[k] * peW1[(size_t)k * HH + t];
  s2[t] = fmaxf(acc, 0.f);
  __syncthreads();
  float z = peb2[t];
  for (int k = 0; k < HH; ++k) z += s2[k] * peW2[(size_t)k * HH + t];
  red[t] = z * z;
  __syncthreads();
  for (int s = 64; s > 0; s >>= 1) { if (t < s) red[t] += red[t + s]; __syncthreads(); }
  out_ze[(size_t)g * HH + t] = z / fmaxf(sqrtf(red[0]), 1e-12f);
  // projection head (fnirs)
  __syncthreads();
  acc = pfb1[t];
  for (int k = 0; k < HH; ++k) acc += shf[k] * pfW1[(size_t)k * HH + t];
  s2[t] = fmaxf(acc, 0.f);
  __syncthreads();
  z = pfb2[t];
  for (int k = 0; k < HH; ++k) z += s2[k] * pfW2[(size_t)k * HH + t];
  red[t] = z * z;
  __syncthreads();
  for (int s = 64; s > 0; s >>= 1) { if (t < s) red[t] += red[t + s]; __syncthreads(); }
  out_zf[(size_t)g * HH + t] = z / fmaxf(sqrtf(red[0]), 1e-12f);
}

// ===========================================================================
extern "C" void kernel_launch(void* const* d_in, const int* in_sizes, int n_in,
                              void* d_out, int out_size, void* d_ws, size_t ws_size,
                              hipStream_t stream) {
  const float* x     = (const float*)d_in[0];
  const int*  eidx   = (const int*)d_in[1];
  const int*  src    = eidx;
  const int*  dstp   = eidx + EE;
  const float* ew    = (const float*)d_in[2];
  const int*  batch  = (const int*)d_in[3];
  const int*  em     = (const int*)d_in[4];
  const float* eeg_W   = (const float*)d_in[6];
  const float* eeg_b   = (const float*)d_in[7];
  const float* fnirs_W = (const float*)d_in[8];
  const float* fnirs_b = (const float*)d_in[9];
  const float* eW1 = (const float*)d_in[10];
  const float* eb1 = (const float*)d_in[11];
  const float* eW2 = (const float*)d_in[12];
  const float* eb2 = (const float*)d_in[13];
  const float* Wi  = (const float*)d_in[14];
  const float* bi  = (const float*)d_in[15];
  const float* Wc  = (const float*)d_in[16];
  const float* bc  = (const float*)d_in[17];
  const float* rel_gate = (const float*)d_in[18];
  const float* bn_g = (const float*)d_in[19];
  const float* bn_b = (const float*)d_in[20];
  const float* coW1 = (const float*)d_in[21];
  const float* cob1 = (const float*)d_in[22];
  const float* coW2 = (const float*)d_in[23];
  const float* cob2 = (const float*)d_in[24];
  const float* gW1  = (const float*)d_in[25];
  const float* gb1  = (const float*)d_in[26];
  const float* gW2  = (const float*)d_in[27];
  const float* gb2  = (const float*)d_in[28];
  const float* cW1  = (const float*)d_in[29];
  const float* cb1  = (const float*)d_in[30];
  const float* cW2  = (const float*)d_in[31];
  const float* cb2  = (const float*)d_in[32];
  const float* peW1 = (const float*)d_in[33];
  const float* peb1 = (const float*)d_in[34];
  const float* peW2 = (const float*)d_in[35];
  const float* peb2 = (const float*)d_in[36];
  const float* pfW1 = (const float*)d_in[37];
  const float* pfb1 = (const float*)d_in[38];
  const float* pfW2 = (const float*)d_in[39];
  const float* pfb2 = (const float*)d_in[40];

  // ---------------- workspace carve-up ----------------
  float* ws = (float*)d_ws;
  size_t off = 0;
  auto A = [&](size_t n) { float* p = ws + off; off += n; return p; };
  float* h    = A((size_t)NN * HH);
  float* xi   = A((size_t)NN * HH);
  float* xc   = A((size_t)NN * HH);
  float* ai   = A((size_t)NN * HH);   // ai & ac contiguous for one memset
  float* ac   = A((size_t)NN * HH);
  float* ewi  = A(EE);
  float* ewc  = A(EE);
  float* din  = A(NN);                // dinv_intra ; dinv_cross contiguous
  float* dcn  = A(NN);
  float* bn   = A(2 * HH);
  float* sg   = A((size_t)GG * HH);   // sums contiguous with counts
  float* se   = A((size_t)GG * HH);
  float* sf   = A((size_t)GG * HH);
  float* cnt  = A(3 * GG);
  off = (off + 7) & ~(size_t)7;       // 32B align for bf16 region
  __bf16* bw     = (__bf16*)(ws + off);
  __bf16* eegWt  = bw;                        // [H][D]
  __bf16* fniWt  = eegWt + (size_t)HH * DD;   // [H][D]
  __bf16* WiT    = fniWt + (size_t)HH * DD;   // [L][H][H]
  __bf16* WcT    = WiT + (size_t)LL * HH * HH;
  __bf16* eW1T   = WcT + (size_t)LL * HH * HH; // [64][H]
  (void)in_sizes; (void)n_in; (void)out_size; (void)ws_size;

  // ---------------- weight prep (fp32 -> transposed bf16) ----------------
  k_t2bf<<<(DD * HH + 255) / 256, 256, 0, stream>>>(eeg_W, eegWt, DD, HH);
  k_t2bf<<<(DD * HH + 255) / 256, 256, 0, stream>>>(fnirs_W, fniWt, DD, HH);
  for (int l = 0; l < LL; ++l) {
    k_t2bf<<<(HH * HH + 255) / 256, 256, 0, stream>>>(Wi + (size_t)l * HH * HH, WiT + (size_t)l * HH * HH, HH, HH);
    k_t2bf<<<(HH * HH + 255) / 256, 256, 0, stream>>>(Wc + (size_t)l * HH * HH, WcT + (size_t)l * HH * HH, HH, HH);
  }
  k_t2bf<<<(HH * 64 + 255) / 256, 256, 0, stream>>>(eW1, eW1T, HH, 64);  // first 128 rows

  // ---------------- pipeline ----------------
  k_proj<<<NN / 16, 256, 0, stream>>>(x, em, eegWt, fniWt, eeg_b, fnirs_b, h);
  k_edge<<<EE / 64, 128, 0, stream>>>(src, dstp, ew, em, h, eW1T, eW1 + 128 * 64,
                                      eb1, eW2, eb2, ewi, ewc);
  k_fill1<<<(2 * NN + 255) / 256, 256, 0, stream>>>(din, 2 * NN);
  k_deg<<<EE / 256, 256, 0, stream>>>(dstp, ewi, ewc, din, dcn);
  k_rsqrt<<<(2 * NN + 255) / 256, 256, 0, stream>>>(din, 2 * NN);

  for (int l = 0; l < LL; ++l) {
    k_xform<<<NN / 16, 256, 0, stream>>>(h, WiT + (size_t)l * HH * HH,
                                         WcT + (size_t)l * HH * HH, xi, xc);
    hipMemsetAsync(ai, 0, (size_t)NN * HH * 2 * sizeof(float), stream);
    hipMemsetAsync(bn, 0, 2 * HH * sizeof(float), stream);
    k_scatter<<<(EE * 32) / 256, 256, 0, stream>>>(src, dstp, ewi, ewc, din, dcn,
                                                   xi, xc, ai, ac);
    k_combine<<<(NN + 63) / 64, 128, 0, stream>>>(xi, xc, ai, ac, din, dcn,
                                                  bi + l * HH, bc + l * HH,
                                                  rel_gate, l, NN, h, bn);
    k_bnfinal<<<1, 128, 0, stream>>>(bn, bn_g + l * HH, bn_b + l * HH, NN);
    k_bnapply<<<(NN * HH) / 256, 256, 0, stream>>>(h, bn);
  }

  hipMemsetAsync(sg, 0, ((size_t)3 * GG * HH + 3 * GG) * sizeof(float), stream);
  k_pool<<<(NN + 63) / 64, 128, 0, stream>>>(h, batch, em, NN, sg, se, sf, cnt);
  k_pool_div<<<GG, 128, 0, stream>>>(sg, se, sf, cnt);

  float* out = (float*)d_out;
  k_head<<<GG, 128, 0, stream>>>(se, sf, sg,
                                 coW1, cob1, coW2, cob2,
                                 gW1, gb1, gW2, gb2,
                                 cW1, cb1, cW2, cb2,
                                 peW1, peb1, peW2, peb2,
                                 pfW1, pfb1, pfW2, pfb2,
                                 out, out + GG * 2, out + GG * 2 + (size_t)GG * HH);
}
